// BigramLanguageModel_89172111000054
// MI455X (gfx1250) — matrix operand order repaired and checked by reading, not compiled, not run on hardware
//
#include <hip/hip_runtime.h>
#include <hip/hip_bf16.h>

// ---------------- model constants ----------------
constexpr int Dm  = 384;
constexpr int Hh  = 6;
constexpr int HSs = 64;
constexpr int Ln  = 5;
constexpr int Vv  = 32000;
constexpr int Tt  = 512;
constexpr int Bb  = 8;
constexpr int FFn = 1536;
constexpr int BT  = Bb * Tt;      // 4096 rows
constexpr int ZB  = Bb * Hh;      // 48 attention batches

typedef __bf16 bf16;
typedef __attribute__((ext_vector_type(16))) __bf16 v16bf;
typedef __attribute__((ext_vector_type(8)))  __bf16 v8bf;
typedef __attribute__((ext_vector_type(8)))  float  v8f;

// epilogue flags
#define FLAG_BIAS 1
#define FLAG_RES  2
#define FLAG_RELU 4
#define FLAG_BF16 8

__device__ __forceinline__ bf16 f2bf(float f) {
    union { float f; unsigned u; } in; in.f = f;
    unsigned r = in.u + 0x7FFFu + ((in.u >> 16) & 1u);   // round-to-nearest-even
    union { unsigned short s; bf16 b; } out; out.s = (unsigned short)(r >> 16);
    return out.b;
}

__device__ __forceinline__ v16bf cat16(v8bf lo, v8bf hi) {
    return __builtin_shufflevector(lo, hi, 0, 1, 2, 3, 4, 5, 6, 7,
                                           8, 9, 10, 11, 12, 13, 14, 15);
}

// =====================================================================
// Batched WMMA GEMM, fully vectorized operand layouts:
//   A: [M][K] bf16, row stride rsA, K contiguous
//   B: "transposed" [N][K] bf16, column-of-C n at stride csB, K contiguous
//   C[m,n] = relu( alpha * sum_k A[m,k]*B[n,k] + bias[n] ) + Res[m,n]
// Wave tile 32x64 (2x4 WMMA tiles, 8 v_wmma per K=32 step).
// Block = 256 threads = 8 waves as 4(M) x 2(N) -> 128x128 block tile.
// Batch z = zb*Hn + zh with per-operand strides.
// =====================================================================
__global__ void __launch_bounds__(256)
k_gemm(const bf16* __restrict__ A, const bf16* __restrict__ B,
       const float* __restrict__ bias, const float* __restrict__ Res,
       void* __restrict__ Cp,
       int M, int N, int K, int Hn,
       long sAb, long sAh, long sBb, long sBh, long sCb, long sCh,
       int rsA, int csB, int ldc, float alpha, int flags)
{
    const int lane = threadIdx.x & 31;
    const int wave = threadIdx.x >> 5;
    const int m0 = blockIdx.y * 128 + (wave >> 1) * 32;
    const int n0 = blockIdx.x * 128 + (wave & 1) * 64;
    if (m0 >= M || n0 >= N) return;

    const int z  = blockIdx.z;
    const int zb = z / Hn, zh = z % Hn;
    const bf16* Az = A + (long)zb * sAb + (long)zh * sAh;
    const bf16* Bz = B + (long)zb * sBb + (long)zh * sBh;
    const long baseC = (long)zb * sCb + (long)zh * sCh;

    const int ml   = lane & 15;
    const int hi8  = (lane & 16) ? 8 : 0;    // C/D row offset per half-wave
    const int hi16 = hi8 << 1;               // B-frag K sub-offset per half-wave

    // A fragment rows m0+ml and m0+16+ml; chunks at +0 and +8 in K
    const bf16* pa0 = Az + (long)(m0 + ml) * rsA + hi16;
    const bf16* pa1 = pa0 + (long)16 * rsA;
    // B fragment columns n0 + {0,16,32,48} + ml; chunks at +0 and +8 in K
    const bf16* pb0 = Bz + (long)(n0 + ml) * csB + hi16;
    const bf16* pb1 = pb0 + (long)16 * csB;
    const bf16* pb2 = pb0 + (long)32 * csB;
    const bf16* pb3 = pb0 + (long)48 * csB;

    v8f acc[2][4] = {};

    for (int k0 = 0; k0 < K; k0 += 32) {
        v16bf a0 = cat16(*(const v8bf*)pa0, *(const v8bf*)(pa0 + 8));
        v16bf a1 = cat16(*(const v8bf*)pa1, *(const v8bf*)(pa1 + 8));
        v16bf b0 = cat16(*(const v8bf*)pb0, *(const v8bf*)(pb0 + 8));
        v16bf b1 = cat16(*(const v8bf*)pb1, *(const v8bf*)(pb1 + 8));
        v16bf b2 = cat16(*(const v8bf*)pb2, *(const v8bf*)(pb2 + 8));
        v16bf b3 = cat16(*(const v8bf*)pb3, *(const v8bf*)(pb3 + 8));
        __builtin_prefetch(pa0 + 64, 0, 1);       // global_prefetch_b8, two K-steps ahead
        __builtin_prefetch(pb0 + 64, 0, 1);
        acc[0][0] = __builtin_amdgcn_wmma_f32_16x16x32_bf16(false, a0, false, b0, (short)0, acc[0][0], false, false);
        acc[0][1] = __builtin_amdgcn_wmma_f32_16x16x32_bf16(false, a0, false, b1, (short)0, acc[0][1], false, false);
        acc[0][2] = __builtin_amdgcn_wmma_f32_16x16x32_bf16(false, a0, false, b2, (short)0, acc[0][2], false, false);
        acc[0][3] = __builtin_amdgcn_wmma_f32_16x16x32_bf16(false, a0, false, b3, (short)0, acc[0][3], false, false);
        acc[1][0] = __builtin_amdgcn_wmma_f32_16x16x32_bf16(false, a1, false, b0, (short)0, acc[1][0], false, false);
        acc[1][1] = __builtin_amdgcn_wmma_f32_16x16x32_bf16(false, a1, false, b1, (short)0, acc[1][1], false, false);
        acc[1][2] = __builtin_amdgcn_wmma_f32_16x16x32_bf16(false, a1, false, b2, (short)0, acc[1][2], false, false);
        acc[1][3] = __builtin_amdgcn_wmma_f32_16x16x32_bf16(false, a1, false, b3, (short)0, acc[1][3], false, false);
        pa0 += 32; pa1 += 32; pb0 += 32; pb1 += 32; pb2 += 32; pb3 += 32;
    }

    // C/D layout: vgpr r -> row m0 + tm*16 + r + (lane>=16 ? 8 : 0), col n0 + tn*16 + (lane&15)
    const int rbase = hi8;
#pragma unroll
    for (int tm = 0; tm < 2; ++tm)
#pragma unroll
        for (int tn = 0; tn < 4; ++tn)
#pragma unroll
            for (int r = 0; r < 8; ++r) {
                const int m = m0 + tm * 16 + rbase + r;
                const int n = n0 + tn * 16 + ml;
                if (m < M && n < N) {
                    float v = acc[tm][tn][r] * alpha;
                    if (flags & FLAG_BIAS) v += bias[n];
                    if (flags & FLAG_RELU) v = fmaxf(v, 0.0f);
                    const long off = baseC + (long)m * ldc + n;
                    if (flags & FLAG_RES)  v += Res[off];
                    if (flags & FLAG_BF16) ((bf16*)Cp)[off] = f2bf(v);
                    else                   ((float*)Cp)[off] = v;
                }
            }
}

// ---------------- embedding ----------------
__global__ void k_embed(const int* __restrict__ x, const float* __restrict__ tok,
                        const float* __restrict__ pos, float* __restrict__ h)
{
    const int row = blockIdx.x;          // b*T + t
    const int i   = threadIdx.x;         // 0..D-1
    const int t   = row % Tt;
    h[(long)row * Dm + i] = tok[(long)x[row] * Dm + i] + pos[(long)t * Dm + i];
}

// ---------------- layernorm (one wave32 per row of 384) ----------------
__global__ void __launch_bounds__(32)
k_ln(const float* __restrict__ x, const float* __restrict__ g,
     const float* __restrict__ b, bf16* __restrict__ out)
{
    const int row  = blockIdx.x;
    const int lane = threadIdx.x;
    const float* xr = x + (long)row * Dm;
    float v[Dm / 32];
    float s = 0.0f;
#pragma unroll
    for (int i = 0; i < Dm / 32; ++i) { v[i] = xr[lane + i * 32]; s += v[i]; }
#pragma unroll
    for (int o = 16; o > 0; o >>= 1) s += __shfl_xor(s, o, 32);
    const float mu = s * (1.0f / Dm);
    float q = 0.0f;
#pragma unroll
    for (int i = 0; i < Dm / 32; ++i) { const float d = v[i] - mu; q += d * d; }
#pragma unroll
    for (int o = 16; o > 0; o >>= 1) q += __shfl_xor(q, o, 32);
    const float rstd = rsqrtf(q * (1.0f / Dm) + 1e-5f);
#pragma unroll
    for (int i = 0; i < Dm / 32; ++i) {
        const int c = lane + i * 32;
        out[(long)row * Dm + c] = f2bf((v[i] - mu) * rstd * g[c] + b[c]);
    }
}

// ---------------- causal softmax (one wave32 per score row of 512) ----------------
__global__ void __launch_bounds__(32)
k_softmax(const float* __restrict__ s, bf16* __restrict__ p)
{
    const long row  = blockIdx.x;        // z*T + t
    const int  t    = (int)(row % Tt);
    const int  lane = threadIdx.x;
    const float* sr = s + row * Tt;
    bf16*       pr = p + row * Tt;
    float vals[Tt / 32];
    float mx = -3.4e38f;
#pragma unroll
    for (int e = 0; e < Tt / 32; ++e) {
        const int j = lane + e * 32;
        const float v = (j <= t) ? sr[j] : -3.4e38f;
        vals[e] = v; mx = fmaxf(mx, v);
    }
#pragma unroll
    for (int o = 16; o > 0; o >>= 1) mx = fmaxf(mx, __shfl_xor(mx, o, 32));
    float sum = 0.0f;
#pragma unroll
    for (int e = 0; e < Tt / 32; ++e) {
        const int j = lane + e * 32;
        const float ev = (j <= t) ? __expf(vals[e] - mx) : 0.0f;
        vals[e] = ev; sum += ev;
    }
#pragma unroll
    for (int o = 16; o > 0; o >>= 1) sum += __shfl_xor(sum, o, 32);
    const float inv = 1.0f / sum;
#pragma unroll
    for (int e = 0; e < Tt / 32; ++e) pr[lane + e * 32] = f2bf(vals[e] * inv);
}

// ---------------- f32 -> bf16 transposing weight converter ----------------
// in: [Lz][K][N] f32 ; out: [Lz][N][K] bf16 (so GEMM B operand is K-contiguous)
__global__ void __launch_bounds__(256)
k_cvtT(const float* __restrict__ in, bf16* __restrict__ out, int K, int N)
{
    __shared__ float tile[32][33];
    const long base = (long)blockIdx.z * K * N;
    const int kt = blockIdx.x * 32, nt = blockIdx.y * 32;
    const int tx = threadIdx.x & 31, ty = threadIdx.x >> 5;   // 32x8
#pragma unroll
    for (int i = ty; i < 32; i += 8) {
        const int k = kt + i, n = nt + tx;
        tile[i][tx] = (k < K && n < N) ? in[base + (long)k * N + n] : 0.0f;
    }
    __syncthreads();
#pragma unroll
    for (int i = ty; i < 32; i += 8) {
        const int n = nt + i, k = kt + tx;
        if (n < N && k < K) out[base + (long)n * K + k] = f2bf(tile[tx][i]);
    }
}

// ---------------- bf16 transpose of v: [B,T,H,HS] -> [B,H,HS,T] ----------------
__global__ void __launch_bounds__(256)
k_trv(const bf16* __restrict__ v, bf16* __restrict__ vT)
{
    __shared__ bf16 tile[32][33];
    const int z = blockIdx.z;                  // b*H + h
    const int b = z / Hh, hh = z % Hh;
    const bf16* src = v  + (long)b * Tt * Dm + (long)hh * HSs;      // [T][HS] stride Dm
    bf16*       dst = vT + (long)b * Dm * Tt + (long)hh * HSs * Tt; // [HS][T] contiguous
    const int tt = blockIdx.x * 32;            // t tile
    const int dt = blockIdx.y * 32;            // d tile
    const int tx = threadIdx.x & 31, ty = threadIdx.x >> 5;
#pragma unroll
    for (int i = ty; i < 32; i += 8)
        tile[i][tx] = src[(long)(tt + i) * Dm + dt + tx];
    __syncthreads();
#pragma unroll
    for (int i = ty; i < 32; i += 8)
        dst[(long)(dt + i) * Tt + tt + tx] = tile[tx][i];
}

// ---------------- host orchestration ----------------
extern "C" void kernel_launch(void* const* d_in, const int* in_sizes, int n_in,
                              void* d_out, int out_size, void* d_ws, size_t ws_size,
                              hipStream_t stream)
{
    (void)in_sizes; (void)n_in; (void)out_size; (void)ws_size;
    const int*   x      = (const int*)  d_in[0];
    const float* tok    = (const float*)d_in[1];
    const float* pos    = (const float*)d_in[2];
    const float* ln1_g  = (const float*)d_in[3];
    const float* ln1_b  = (const float*)d_in[4];
    const float* Wq     = (const float*)d_in[5];
    const float* Wk     = (const float*)d_in[6];
    const float* Wv     = (const float*)d_in[7];
    const float* Wp     = (const float*)d_in[8];
    const float* bp     = (const float*)d_in[9];
    const float* ln2_g  = (const float*)d_in[10];
    const float* ln2_b  = (const float*)d_in[11];
    const float* W1     = (const float*)d_in[12];
    const float* b1     = (const float*)d_in[13];
    const float* W2     = (const float*)d_in[14];
    const float* b2     = (const float*)d_in[15];
    const float* lnf_g  = (const float*)d_in[16];
    const float* lnf_b  = (const float*)d_in[17];
    const float* Wout   = (const float*)d_in[18];
    const float* bout   = (const float*)d_in[19];

    // ---- workspace carve (256B aligned) ----
    char* wp = (char*)d_ws;
    auto alloc = [&](size_t bytes) -> void* {
        void* r = (void*)wp;
        wp += (bytes + 255) & ~size_t(255);
        return r;
    };
    bf16* WqT = (bf16*)alloc((size_t)Ln * Dm * Dm  * 2);   // [l][N=D][K=D]
    bf16* WkT = (bf16*)alloc((size_t)Ln * Dm * Dm  * 2);
    bf16* WvT = (bf16*)alloc((size_t)Ln * Dm * Dm  * 2);
    bf16* WpT = (bf16*)alloc((size_t)Ln * Dm * Dm  * 2);
    bf16* W1T = (bf16*)alloc((size_t)Ln * Dm * FFn * 2);   // [l][FF][D]
    bf16* W2T = (bf16*)alloc((size_t)Ln * FFn * Dm * 2);   // [l][D][FF]
    bf16* WoT = (bf16*)alloc((size_t)Dm * Vv * 2);         // [V][D]
    float* h   = (float*)alloc((size_t)BT * Dm * 4);
    bf16* abf  = (bf16*)alloc((size_t)BT * Dm * 2);
    bf16* qbf  = (bf16*)alloc((size_t)BT * Dm * 2);
    bf16* kbf  = (bf16*)alloc((size_t)BT * Dm * 2);
    bf16* vbf  = (bf16*)alloc((size_t)BT * Dm * 2);
    bf16* vTbf = (bf16*)alloc((size_t)BT * Dm * 2);        // [B,H,HS,T]
    bf16* obf  = (bf16*)alloc((size_t)BT * Dm * 2);
    float* scr = (float*)alloc((size_t)ZB * Tt * Tt * 4);
    bf16* pbf  = (bf16*)alloc((size_t)ZB * Tt * Tt * 2);
    bf16* f1bf = (bf16*)alloc((size_t)BT * FFn * 2);

    // ---- transpose+convert weights to bf16 [N][K] ----
    auto cvtT = [&](const float* in, bf16* out, int K, int N, int layers) {
        dim3 g(K / 32, N / 32, layers);
        k_cvtT<<<g, dim3(256), 0, stream>>>(in, out, K, N);
    };
    cvtT(Wq, WqT, Dm, Dm, Ln);
    cvtT(Wk, WkT, Dm, Dm, Ln);
    cvtT(Wv, WvT, Dm, Dm, Ln);
    cvtT(Wp, WpT, Dm, Dm, Ln);
    cvtT(W1, W1T, Dm, FFn, Ln);
    cvtT(W2, W2T, FFn, Dm, Ln);
    cvtT(Wout, WoT, Dm, Vv, 1);

    // ---- embedding ----
    k_embed<<<dim3(BT), dim3(Dm), 0, stream>>>(x, tok, pos, h);

    auto gemm = [&](const bf16* A, const bf16* Bw, const float* bias, const float* Res,
                    void* C, int M, int N, int K, int Hn, int Zn,
                    long sAb, long sAh, long sBb, long sBh, long sCb, long sCh,
                    int rsA, int csB, int ldc, float alpha, int flags) {
        dim3 g((N + 127) / 128, (M + 127) / 128, Zn);
        k_gemm<<<g, dim3(256), 0, stream>>>(A, Bw, bias, Res, C, M, N, K, Hn,
                                            sAb, sAh, sBb, sBh, sCb, sCh,
                                            rsA, csB, ldc, alpha, flags);
    };

    const float scale = 0.05103103630798288f;   // D^-0.5 (reference scales by n_embed)
    const long TD = (long)Tt * Dm;
    const long TT = (long)Tt * Tt;

    for (int l = 0; l < Ln; ++l) {
        const long wOff  = (long)l * Dm * Dm;
        const long w1Off = (long)l * Dm * FFn;

        // LN1
        k_ln<<<dim3(BT), dim3(32), 0, stream>>>(h, ln1_g + l * Dm, ln1_b + l * Dm, abf);

        // q, k, v projections (no bias), bf16 outputs
        gemm(abf, WqT + wOff, nullptr, nullptr, qbf, BT, Dm, Dm, 1, 1,
             0, 0, 0, 0, 0, 0, Dm, Dm, Dm, 1.0f, FLAG_BF16);
        gemm(abf, WkT + wOff, nullptr, nullptr, kbf, BT, Dm, Dm, 1, 1,
             0, 0, 0, 0, 0, 0, Dm, Dm, Dm, 1.0f, FLAG_BF16);
        gemm(abf, WvT + wOff, nullptr, nullptr, vbf, BT, Dm, Dm, 1, 1,
             0, 0, 0, 0, 0, 0, Dm, Dm, Dm, 1.0f, FLAG_BF16);

        // vT[b,h,d,t] = v[b,t,h,d]
        k_trv<<<dim3(Tt / 32, HSs / 32, ZB), dim3(256), 0, stream>>>(vbf, vTbf);

        // scores[z] = q[b,:,h,:] @ k[b,:,h,:]^T * scale  (k rows ARE K-contiguous columns)
        gemm(qbf, kbf, nullptr, nullptr, scr, Tt, Tt, HSs, Hh, ZB,
             TD, HSs, TD, HSs, (long)Hh * TT, TT,
             Dm, Dm, Tt, scale, 0);

        // causal softmax -> bf16 probs
        k_softmax<<<dim3(ZB * Tt), dim3(32), 0, stream>>>(scr, pbf);

        // o[z] = probs @ v  (B = vT, K(=t)-contiguous)
        gemm(pbf, vTbf, nullptr, nullptr, obf, Tt, HSs, Tt, Hh, ZB,
             (long)Hh * TT, TT, (long)Dm * Tt, (long)HSs * Tt, TD, HSs,
             Tt, Tt, Dm, 1.0f, FLAG_BF16);

        // h = h + o @ Wproj + bproj
        gemm(obf, WpT + wOff, bp + l * Dm, h, h, BT, Dm, Dm, 1, 1,
             0, 0, 0, 0, 0, 0, Dm, Dm, Dm, 1.0f, FLAG_BIAS | FLAG_RES);

        // LN2
        k_ln<<<dim3(BT), dim3(32), 0, stream>>>(h, ln2_g + l * Dm, ln2_b + l * Dm, abf);

        // f1 = relu(a @ W1 + b1) -> bf16
        gemm(abf, W1T + w1Off, b1 + (long)l * FFn, nullptr, f1bf, BT, FFn, Dm, 1, 1,
             0, 0, 0, 0, 0, 0, Dm, Dm, FFn, 1.0f, FLAG_BIAS | FLAG_RELU | FLAG_BF16);

        // h = h + f1 @ W2 + b2
        gemm(f1bf, W2T + w1Off, b2 + l * Dm, h, h, BT, Dm, FFn, 1, 1,
             0, 0, 0, 0, 0, 0, FFn, FFn, Dm, 1.0f, FLAG_BIAS | FLAG_RES);
    }

    // final LN + logits
    k_ln<<<dim3(BT), dim3(32), 0, stream>>>(h, lnf_g, lnf_b, abf);
    gemm(abf, WoT, bout, nullptr, d_out, BT, Vv, Dm, 1, 1,
         0, 0, 0, 0, 0, 0, Dm, Dm, Vv, 1.0f, FLAG_BIAS);
}
